// FullyConnectedTree_69904887710210
// MI455X (gfx1250) — matrix operand order repaired and checked by reading, not compiled, not run on hardware
//
#include <hip/hip_runtime.h>

typedef __bf16 bf16;
typedef __attribute__((ext_vector_type(16))) __bf16 v16bf;
typedef __attribute__((ext_vector_type(8)))  __bf16 v8bf;
typedef __attribute__((ext_vector_type(4)))  __bf16 v4bf;
typedef __attribute__((ext_vector_type(8)))  float  v8f;
typedef __attribute__((ext_vector_type(4)))  float  v4f;

#define D_DIM   1024
#define B_ROWS  16384
#define NLAYERS 5

__device__ __forceinline__ float sigmoidf_(float x) {
    return 1.0f / (1.0f + __expf(-x));
}

// CDNA5 async global->LDS copy of 16 bytes per lane (ASYNCcnt-tracked).
__device__ __forceinline__ void async_cp16(unsigned lds_off, const bf16* g) {
    unsigned long long ga = (unsigned long long)(size_t)g;
    asm volatile("global_load_async_to_lds_b128 %0, %1, off"
                 :: "v"(lds_off), "v"(ga)
                 : "memory");
}

// Wait for this wave's async copies, then split workgroup barrier.
// (Avoids __syncthreads' global-memory fence, which drains loadcnt.)
__device__ __forceinline__ void wg_barrier_async() {
    asm volatile("s_wait_asynccnt 0x0" ::: "memory");
    asm volatile("s_barrier_signal -1\n\ts_barrier_wait -1" ::: "memory");
}

// ---------------------------------------------------------------------------
// Prep: W_t[l][n][k] = bf16( sigmoid(logits[l][k][n]/3) * scales[l][k][n] )
// 32x32 LDS-tiled transpose, coalesced reads and writes.
// grid (32, 32, 5), block 256
// ---------------------------------------------------------------------------
__global__ void prep_weights_kernel(const float* __restrict__ logits,
                                    const float* __restrict__ scales,
                                    bf16* __restrict__ wt) {
    __shared__ float tile[32][33];
    const int l  = blockIdx.z;
    const int n0 = blockIdx.x * 32;
    const int k0 = blockIdx.y * 32;
    const int tx = threadIdx.x & 31;
    const int ty = threadIdx.x >> 5;   // 0..7
#pragma unroll
    for (int i = 0; i < 4; ++i) {
        int k = k0 + ty + i * 8;
        int n = n0 + tx;
        size_t idx = ((size_t)l * D_DIM + k) * D_DIM + n;
        tile[ty + i * 8][tx] = sigmoidf_(logits[idx] * (1.0f / 3.0f)) * scales[idx];
    }
    __syncthreads();
#pragma unroll
    for (int i = 0; i < 4; ++i) {
        int n = n0 + ty + i * 8;
        int k = k0 + tx;
        wt[((size_t)l * D_DIM + n) * D_DIM + k] = (bf16)tile[tx][ty + i * 8];
    }
}

// ---------------------------------------------------------------------------
// Prep: per-layer andness biases, output weights, output bias.  1 block.
// ---------------------------------------------------------------------------
__global__ void prep_bias_kernel(const float* __restrict__ biases_h, // [5][1024]
                                 const float* __restrict__ logits_o, // [1024]
                                 const float* __restrict__ scales_o, // [1024]
                                 const float* __restrict__ bias_o,   // [1]
                                 float* __restrict__ abias,          // [5][1024]
                                 float* __restrict__ wout,           // [1024]
                                 float* __restrict__ aout) {         // [1]
    int t = threadIdx.x;
    for (int j = t; j < NLAYERS * D_DIM; j += blockDim.x)
        abias[j] = sigmoidf_(biases_h[j]) * 3.0f - 1.0f;
    if (t < D_DIM)
        wout[t] = sigmoidf_(logits_o[t] * (1.0f / 3.0f)) * scales_o[t];
    if (t == 0)
        aout[0] = sigmoidf_(bias_o[0]) * 3.0f - 1.0f;
}

// ---------------------------------------------------------------------------
// x (f32) -> bf16, 4 elements/thread.  grid 16384, block 256 (exact cover)
// ---------------------------------------------------------------------------
__global__ void x_to_bf16_kernel(const float* __restrict__ x, bf16* __restrict__ o) {
    size_t i = ((size_t)blockIdx.x * blockDim.x + threadIdx.x) * 4;
    v4f v = *(const v4f*)(x + i);
    v4bf h;
    h.x = (bf16)v.x; h.y = (bf16)v.y; h.z = (bf16)v.z; h.w = (bf16)v.w;
    *(v4bf*)(o + i) = h;
}

// ---------------------------------------------------------------------------
// GEMM: C[M][D] = bf16( A[M][D] @ W + bias ), W given transposed as Bt[n][k].
// Block tile 256(M) x 128(N), BK=32.  8 waves as 4(M) x 2(N); each wave owns a
// 64x64 output tile = 4x4 v_wmma_f32_16x16x32_bf16 accumulators (128 VGPRs).
// Global->LDS staging uses CDNA5 GLOBAL_LOAD_ASYNC_TO_LDS_B128 (ASYNCcnt),
// double-buffered, one s_wait_asynccnt + split barrier per K-step.
// grid (8, 64), block 256
// ---------------------------------------------------------------------------
#define BM 256
#define BN 128
#define BK 32
#define LSTR 40                     // padded LDS row stride (bf16 elems)
#define NKITER (D_DIM / BK)         // 32

__global__ __launch_bounds__(256)
void gemm_layer_kernel(const bf16* __restrict__ A,     // [M][D]
                       const bf16* __restrict__ Bt,    // [D][D]  ([n][k])
                       const float* __restrict__ bias, // [D]
                       bf16* __restrict__ Co) {        // [M][D]
    __shared__ alignas(16) bf16 As[2][BM * LSTR];      // 2 x 20 KB
    __shared__ alignas(16) bf16 Bs[2][BN * LSTR];      // 2 x 10 KB

    const int tid  = threadIdx.x;
    const int lane = tid & 31;
    const int wave = tid >> 5;
    const int hlf  = lane >> 4;     // lane half: K-split per WMMA layout
    const int l16  = lane & 15;
    const int wm   = wave >> 1;     // 0..3  (M)
    const int wn   = wave & 1;      // 0..1  (N)

    const int m0 = blockIdx.y * BM;
    const int n0 = blockIdx.x * BN;

    // Copy mapping: 16B chunks; A tile 1024 chunks (4/thread), B tile 512 (2/thread)
    const int r0 = tid >> 2;            // 0..63
    const int c8 = (tid & 3) * 8;       // element offset within 32-wide K slab
    const bf16* aP[4];
    int sA[4];
#pragma unroll
    for (int t = 0; t < 4; ++t) {
        aP[t] = A + ((size_t)(m0 + r0 + t * 64) * D_DIM) + c8;
        sA[t] = (r0 + t * 64) * LSTR + c8;
    }
    const bf16* bP0 = Bt + ((size_t)(n0 + r0     ) * D_DIM) + c8;
    const bf16* bP1 = Bt + ((size_t)(n0 + r0 + 64) * D_DIM) + c8;
    const int sB0 = r0 * LSTR + c8;
    const int sB1 = (r0 + 64) * LSTR + c8;

    v8f acc[4][4] = {};

    // prologue: stage 0 -> buffer 0
    {
        const int ko = 0;
#pragma unroll
        for (int t = 0; t < 4; ++t)
            async_cp16((unsigned)(size_t)&As[0][sA[t]], aP[t] + ko);
        async_cp16((unsigned)(size_t)&Bs[0][sB0], bP0 + ko);
        async_cp16((unsigned)(size_t)&Bs[0][sB1], bP1 + ko);
    }
    wg_barrier_async();

#pragma unroll 2
    for (int kt = 0; kt < NKITER; ++kt) {
        const int buf = kt & 1;
        if (kt + 1 < NKITER) {          // async-prefetch next slab into buf^1
            const int ko = (kt + 1) * BK;
#pragma unroll
            for (int t = 0; t < 4; ++t)
                async_cp16((unsigned)(size_t)&As[buf ^ 1][sA[t]], aP[t] + ko);
            async_cp16((unsigned)(size_t)&Bs[buf ^ 1][sB0], bP0 + ko);
            async_cp16((unsigned)(size_t)&Bs[buf ^ 1][sB1], bP1 + ko);
        }

        // B fragments: lane = column n (l16), K = 16*hlf .. 16*hlf+15 contiguous
        union Frag { v16bf v; v8bf h[2]; };
        Frag bfrag[4], afrag;
#pragma unroll
        for (int ns = 0; ns < 4; ++ns) {
            const bf16* p = &Bs[buf][(wn * 64 + ns * 16 + l16) * LSTR + hlf * 16];
            bfrag[ns].h[0] = *(const v8bf*)(p);
            bfrag[ns].h[1] = *(const v8bf*)(p + 8);
        }
        // A fragments: lane = row m (l16), K = 16*hlf .. 16*hlf+15 contiguous
#pragma unroll
        for (int ms = 0; ms < 4; ++ms) {
            const bf16* p = &As[buf][(wm * 64 + ms * 16 + l16) * LSTR + hlf * 16];
            afrag.h[0] = *(const v8bf*)(p);
            afrag.h[1] = *(const v8bf*)(p + 8);
#pragma unroll
            for (int ns = 0; ns < 4; ++ns) {
                acc[ms][ns] = __builtin_amdgcn_wmma_f32_16x16x32_bf16(
                    false, afrag.v, false, bfrag[ns].v,
                    (short)0, acc[ms][ns], false, false);
            }
        }
        // our ds reads were consumed (dscnt-waited before wmma); async copies for
        // buf^1 complete here; then all waves rendezvous.
        wg_barrier_async();
    }

    // Epilogue: D layout — element r: M = 8*hlf + r, N = l16
    const int nb0 = n0 + wn * 64;
    float ab[4];
#pragma unroll
    for (int ns = 0; ns < 4; ++ns)
        ab[ns] = bias[nb0 + ns * 16 + l16];
#pragma unroll
    for (int ms = 0; ms < 4; ++ms) {
        const int row = m0 + wm * 64 + ms * 16 + hlf * 8;
#pragma unroll
        for (int ns = 0; ns < 4; ++ns) {
            const int col = nb0 + ns * 16 + l16;
#pragma unroll
            for (int r = 0; r < 8; ++r)
                Co[(size_t)(row + r) * D_DIM + col] = (bf16)(acc[ms][ns][r] + ab[ns]);
        }
    }
}

// ---------------------------------------------------------------------------
// Output GEMV: one wave per row, vectorized v8bf loads.  grid 2048, block 256
// ---------------------------------------------------------------------------
__global__ void gemv_out_kernel(const bf16* __restrict__ act,
                                const float* __restrict__ wout,
                                const float* __restrict__ aout,
                                float* __restrict__ out) {
    const int gid  = blockIdx.x * blockDim.x + threadIdx.x;
    const int row  = gid >> 5;
    const int lane = gid & 31;
    const bf16* p = act + (size_t)row * D_DIM;
    float s = 0.0f;
#pragma unroll
    for (int t = 0; t < 4; ++t) {
        const int k = t * 256 + lane * 8;
        v8bf a = *(const v8bf*)(p + k);
#pragma unroll
        for (int j = 0; j < 8; ++j)
            s += (float)a[j] * wout[k + j];
    }
#pragma unroll
    for (int off = 16; off > 0; off >>= 1)
        s += __shfl_xor(s, off, 32);
    if (lane == 0) out[row] = s + aout[0];
}

// ---------------------------------------------------------------------------
extern "C" void kernel_launch(void* const* d_in, const int* in_sizes, int n_in,
                              void* d_out, int out_size, void* d_ws, size_t ws_size,
                              hipStream_t stream) {
    (void)in_sizes; (void)n_in; (void)out_size; (void)ws_size;
    const float* x        = (const float*)d_in[0];
    const float* logits_h = (const float*)d_in[1];
    const float* scales_h = (const float*)d_in[2];
    const float* biases_h = (const float*)d_in[3];
    const float* logits_o = (const float*)d_in[4];
    const float* scales_o = (const float*)d_in[5];
    const float* bias_o   = (const float*)d_in[6];
    float* out = (float*)d_out;

    char* ws = (char*)d_ws;
    size_t off = 0;
    bf16* wt = (bf16*)(ws + off);        off += (size_t)NLAYERS * D_DIM * D_DIM * sizeof(bf16);
    float* abias = (float*)(ws + off);   off += (size_t)NLAYERS * D_DIM * sizeof(float);
    float* wout  = (float*)(ws + off);   off += (size_t)D_DIM * sizeof(float);
    float* aout  = (float*)(ws + off);   off += 16;
    bf16* actA = (bf16*)(ws + off);      off += (size_t)B_ROWS * D_DIM * sizeof(bf16);
    bf16* actB = (bf16*)(ws + off);

    prep_weights_kernel<<<dim3(D_DIM / 32, D_DIM / 32, NLAYERS), 256, 0, stream>>>(
        logits_h, scales_h, wt);
    prep_bias_kernel<<<1, 1024, 0, stream>>>(biases_h, logits_o, scales_o, bias_o,
                                             abias, wout, aout);
    x_to_bf16_kernel<<<(B_ROWS * D_DIM) / (256 * 4), 256, 0, stream>>>(x, actA);

    const bf16* cur = actA;
    bf16* nxt = actB;
    for (int l = 0; l < NLAYERS; ++l) {
        gemm_layer_kernel<<<dim3(D_DIM / BN, B_ROWS / BM), 256, 0, stream>>>(
            cur, wt + (size_t)l * D_DIM * D_DIM, abias + (size_t)l * D_DIM, nxt);
        bf16* t = (bf16*)cur; cur = nxt; nxt = t;
    }
    gemv_out_kernel<<<(B_ROWS * 32) / 256, 256, 0, stream>>>(cur, wout, aout, out);
}
